// IntrinsicSignalSynthesizer_48292612276361
// MI455X (gfx1250) — compile-verified, matrix operand order checked by reading, not run on hardware
//
#include <hip/hip_runtime.h>
#include <hip/hip_bf16.h>
#include <math.h>
#include <stdint.h>

// MI455X / gfx1250: wave32, WMMA 16x16x32 bf16 -> f32 accumulate,
// TDM (tensor_load_to_lds) for activation-tile staging into LDS.

typedef __bf16 bf16;
typedef __attribute__((ext_vector_type(16))) __bf16 v16bf;
typedef __attribute__((ext_vector_type(8)))  float  v8f;

#define BROWS 16384
#define DDIM  1024
#define MROWS 100
#define MPAD  112

// ---------------------------------------------------------------- TDM glue
#if defined(__has_builtin)
#if __has_builtin(__builtin_amdgcn_tensor_load_to_lds)
#define HAVE_TDM 1
#endif
#endif
#ifndef HAVE_TDM
#define HAVE_TDM 0
#endif

#if __has_include(<hip/amd_detail/amd_gfx1250_TDM.h>)
#define TDM_SIX_ARGS 1
#endif

__device__ __forceinline__ unsigned lds_offset(const void* p) {
  // generic LDS address keeps the LDS byte offset in the low 32 bits
  return (unsigned)(unsigned long long)(uintptr_t)p;
}

#if HAVE_TDM
typedef unsigned int u32x4 __attribute__((ext_vector_type(4)));
typedef int          i32x4 __attribute__((ext_vector_type(4)));
typedef int          i32x8 __attribute__((ext_vector_type(8)));

// 2D tile load: tileRows x tileCols of 2-byte elements, row stride in elements.
// D# layout per CDNA5 ISA ch.8 (group0: count/lds/global/type, group1: dims).
__device__ __forceinline__ void tdm_load_2d(const void* gsrc, unsigned ldsaddr,
                                            unsigned tileRows, unsigned tileCols,
                                            unsigned long long rowStrideElems) {
  unsigned long long ga = (unsigned long long)(uintptr_t)gsrc;
  u32x4 g0 = {};
  g0[0] = 1u;                                                   // count=1, user D#
  g0[1] = ldsaddr;                                              // lds_addr
  g0[2] = (unsigned)(ga & 0xffffffffull);                       // global_addr[31:0]
  g0[3] = (unsigned)((ga >> 32) & 0x01ffffffull) | (2u << 30);  // [56:32], type=2
  const unsigned td0 = tileCols, td1 = tileRows;
  const unsigned long long s0 = rowStrideElems;
  i32x8 g1 = {};
  g1[0] = (int)(1u << 16);                                // data_size=1 (2B), wg_mask=0
  g1[1] = (int)((td0 & 0xffffu) << 16);                   // tensor_dim0[15:0] @bit48
  g1[2] = (int)((td0 >> 16) | ((td1 & 0xffffu) << 16));   // dim0 hi | dim1 lo
  g1[3] = (int)((td1 >> 16) | ((tileCols & 0xffffu) << 16)); // dim1 hi | tile_dim0
  g1[4] = (int)(tileRows & 0xffffu);                      // tile_dim1 (tile_dim2=0)
  g1[5] = (int)(s0 & 0xffffffffull);                      // dim0_stride[31:0]
  g1[6] = (int)((s0 >> 32) & 0xffffull);                  // dim0_stride[47:32]
  g1[7] = 0;
  i32x4 g2 = {};
  i32x4 g3 = {};
#if defined(TDM_SIX_ARGS)
  i32x8 g4 = {};
  __builtin_amdgcn_tensor_load_to_lds(g0, g1, g2, g3, g4, 0);
#else
  __builtin_amdgcn_tensor_load_to_lds(g0, g1, g2, g3, 0);
#endif
}
#endif  // HAVE_TDM

__device__ __forceinline__ void wait_tensor0() {
#if defined(__has_builtin) && __has_builtin(__builtin_amdgcn_s_wait_tensorcnt)
  __builtin_amdgcn_s_wait_tensorcnt(0);
#else
  asm volatile("s_wait_tensorcnt 0x0" ::: "memory");
#endif
}

// fallback cooperative staging (32B chunks; cols % 16 == 0)
__device__ __forceinline__ void stage_tile(const bf16* __restrict__ g,
                                           bf16* __restrict__ l,
                                           int rows, int cols, int rstride,
                                           int tid, int nthreads) {
  const int total = rows * cols;
  for (int i = tid * 16; i < total; i += nthreads * 16) {
    const int r = i / cols, c = i - r * cols;
    *(v16bf*)(l + i) = *(const v16bf*)(g + (size_t)r * rstride + c);
  }
}

// ---------------------------------------------------------------- helpers
__device__ __forceinline__ v8f wmma_bf16(v16bf a, v16bf b, v8f c) {
  return __builtin_amdgcn_wmma_f32_16x16x32_bf16(false, a, false, b,
                                                 (short)0, c, false, false);
}

__device__ __forceinline__ float softplus_f(float x) {
  return x > 20.0f ? x : log1pf(__expf(x));
}

// ---------------------------------------------------------------- converts
__global__ void cvt_f32_to_bf16(const float* __restrict__ src,
                                bf16* __restrict__ dst, int n) {
  int i = blockIdx.x * blockDim.x + threadIdx.x;
  if (i < n) dst[i] = (bf16)src[i];
}

__global__ void cvt_pad_mem(const float* __restrict__ src,
                            bf16* __restrict__ dst, int rows, int cols, int padRows) {
  int i = blockIdx.x * blockDim.x + threadIdx.x;
  int n = padRows * cols;
  if (i >= n) return;
  int r = i / cols;
  dst[i] = (r < rows) ? (bf16)src[i] : (bf16)0.0f;
}

// ---------------------------------------------------------------- m_norm
__global__ void mnorm_kernel(const float* __restrict__ mem,
                             float* __restrict__ mnorm, int rows, int cols, int padRows) {
  int r = blockIdx.x * blockDim.x + threadIdx.x;
  if (r >= padRows) return;
  float ss = 0.0f;
  if (r < rows) {
    const float* p = mem + (size_t)r * cols;
    for (int c = 0; c < cols; ++c) { float v = p[c]; ss += v * v; }
  }
  mnorm[r] = fmaxf(sqrtf(ss), 1e-8f);
}

// ------------------------------------------------- per-row entropy + |actual|
__global__ void row_stats_kernel(const float* __restrict__ pred,
                                 const float* __restrict__ act,
                                 float* __restrict__ entropy,
                                 float* __restrict__ a_norm, int D) {
  const int row = blockIdx.x;
  const int t = threadIdx.x;
  const float* pr = pred + (size_t)row * D;
  const float* ar = act  + (size_t)row * D;
  float x[4];
  float lm = -1e30f;
  for (int j = 0; j < 4; ++j) { x[j] = pr[t + 256 * j]; lm = fmaxf(lm, x[j]); }
  __shared__ float red[256];
  red[t] = lm; __syncthreads();
  for (int s = 128; s > 0; s >>= 1) {
    if (t < s) red[t] = fmaxf(red[t], red[t + s]);
    __syncthreads();
  }
  const float m = red[0];
  __syncthreads();
  float S = 0.0f, T = 0.0f, ss = 0.0f;
  for (int j = 0; j < 4; ++j) {
    float e = __expf(x[j] - m);
    S += e; T += x[j] * e;
    float a = ar[t + 256 * j]; ss += a * a;
  }
  red[t] = S; __syncthreads();
  for (int s = 128; s > 0; s >>= 1) { if (t < s) red[t] += red[t + s]; __syncthreads(); }
  S = red[0]; __syncthreads();
  red[t] = T; __syncthreads();
  for (int s = 128; s > 0; s >>= 1) { if (t < s) red[t] += red[t + s]; __syncthreads(); }
  T = red[0]; __syncthreads();
  red[t] = ss; __syncthreads();
  for (int s = 128; s > 0; s >>= 1) { if (t < s) red[t] += red[t + s]; __syncthreads(); }
  if (t == 0) {
    entropy[row] = logf(S) + m - T / S;
    a_norm[row]  = fmaxf(sqrtf(red[0]), 1e-8f);
  }
}

// -------------------------------------------- fused 2-layer MLP -> softplus
// Activation tile (16 x Ktot) staged into LDS via TDM; A-frags via ds_load,
// B-frags (weights) stream from L2. Concat handled as two wave-uniform K loops.
__global__ void fused_mlp1_kernel(const bf16* __restrict__ Xa,
                                  const bf16* __restrict__ Xb,
                                  const bf16* __restrict__ W1,   // [N][Ka+Kb]
                                  const float* __restrict__ b1,  // [N]
                                  const float* __restrict__ w2,  // [N]
                                  const float* __restrict__ b2,  // [1]
                                  float* __restrict__ out,
                                  int N, int Ka, int Kb) {
  extern __shared__ __align__(32) bf16 smem[];   // 16 * (Ka + Kb)
  const int lane = threadIdx.x & 31;
  const int wave = threadIdx.x >> 5;   // 0..7
  const int half = lane >> 4;
  const int ln   = lane & 15;
  const int row0 = blockIdx.x * 16;
  const int Ktot = Ka + Kb;
  bf16* tA = smem;
  bf16* tB = smem + 16 * Ka;

  __shared__ float rowacc[16];
  if (threadIdx.x < 16) rowacc[threadIdx.x] = 0.0f;

#if HAVE_TDM
  if (wave == 0) {
    tdm_load_2d(Xa + (size_t)row0 * Ka, lds_offset(tA), 16, Ka, Ka);
    if (Kb > 0)
      tdm_load_2d(Xb + (size_t)row0 * Kb, lds_offset(tB), 16, Kb, Kb);
    wait_tensor0();
  }
#else
  stage_tile(Xa + (size_t)row0 * Ka, tA, 16, Ka, Ka, threadIdx.x, blockDim.x);
  if (Kb > 0)
    stage_tile(Xb + (size_t)row0 * Kb, tB, 16, Kb, Kb, threadIdx.x, blockDim.x);
#endif
  __syncthreads();

  float part[8];
#pragma unroll
  for (int r = 0; r < 8; ++r) part[r] = 0.0f;

  const int Ntiles = N >> 4;
  for (int nt = wave; nt < Ntiles; nt += 8) {
    const int n0 = nt * 16;
    const bf16* wrow = W1 + (size_t)(n0 + ln) * Ktot;
    v8f acc = {};
    for (int k = 0; k < Ka; k += 32) {
      v16bf a = *(const v16bf*)(tA + ln * Ka + k + half * 16);     // ds_load
      v16bf b = *(const v16bf*)(wrow + k + half * 16);
      __builtin_prefetch(wrow + k + 128, 0, 0);                    // global_prefetch_b8
      acc = wmma_bf16(a, b, acc);
    }
    for (int k = 0; k < Kb; k += 32) {
      v16bf a = *(const v16bf*)(tB + ln * Kb + k + half * 16);
      v16bf b = *(const v16bf*)(wrow + Ka + k + half * 16);
      acc = wmma_bf16(a, b, acc);
    }
    const int n = n0 + ln;
    const float w2n = w2[n];
    const float b1n = b1[n];
#pragma unroll
    for (int r = 0; r < 8; ++r) {
      float v = acc[r] + b1n;          // C/D layout: half h holds rows h*8+r
      v = v > 0.0f ? v : 0.0f;
      part[r] += v * w2n;
    }
  }
#pragma unroll
  for (int r = 0; r < 8; ++r)
    atomicAdd(&rowacc[half * 8 + r], part[r]);   // ds_add_f32
  __syncthreads();
  if (threadIdx.x < 16)
    out[row0 + threadIdx.x] = softplus_f(rowacc[threadIdx.x] + b2[0]);
}

// ------------------------------------------------------ cosine-sim max (WMMA)
__global__ void novelty_sims_kernel(const bf16* __restrict__ Act,   // [B][K]
                                    const bf16* __restrict__ Mem,   // [MPAD][K]
                                    const float* __restrict__ a_norm,
                                    const float* __restrict__ m_norm,
                                    float* __restrict__ simmax, int K) {
  extern __shared__ __align__(32) bf16 smem[];   // 16 * K act tile
  const int lane = threadIdx.x & 31;
  const int wave = threadIdx.x >> 5;
  const int half = lane >> 4;
  const int ln   = lane & 15;
  const int row0 = blockIdx.x * 16;

  __shared__ float wmax[8][16];
  if (threadIdx.x < 128) ((float*)wmax)[threadIdx.x] = -1e30f;

#if HAVE_TDM
  if (wave == 0) {
    tdm_load_2d(Act + (size_t)row0 * K, lds_offset(smem), 16, K, K);
    wait_tensor0();
  }
#else
  stage_tile(Act + (size_t)row0 * K, smem, 16, K, K, threadIdx.x, blockDim.x);
#endif
  __syncthreads();

  const int Ntiles = MPAD >> 4;  // 7
  if (wave < Ntiles) {
    const int n0 = wave * 16;
    v8f acc = {};
    for (int k = 0; k < K; k += 32) {
      v16bf a = *(const v16bf*)(smem + ln * K + k + half * 16);    // ds_load
      v16bf b = *(const v16bf*)(Mem + (size_t)(n0 + ln) * K + k + half * 16);
      acc = wmma_bf16(a, b, acc);
    }
    const int n = n0 + ln;
    const float inv = 1.0f / m_norm[n];
    float lmax[8];
#pragma unroll
    for (int r = 0; r < 8; ++r) {
      float s = (n < MROWS) ? acc[r] * inv : -1e30f;
      for (int m = 1; m < 16; m <<= 1) s = fmaxf(s, __shfl_xor(s, m, 32));
      lmax[r] = s;
    }
    if (ln == 0) {
#pragma unroll
      for (int r = 0; r < 8; ++r) wmax[wave][half * 8 + r] = lmax[r];
    }
  }
  __syncthreads();
  if (threadIdx.x < 16) {
    float mx = -1e30f;
    for (int w = 0; w < 8; ++w) mx = fmaxf(mx, wmax[w][threadIdx.x]);
    simmax[row0 + threadIdx.x] = mx / a_norm[row0 + threadIdx.x];
  }
}

// --------------------------------------- autoencoder MSE (two fused GEMMs)
__global__ void compression_kernel(const float* __restrict__ Pred,     // f32 [B][D]
                                   const bf16* __restrict__ PredBf,    // [B][D]
                                   const bf16* __restrict__ Wc1,       // [H][D]
                                   const float* __restrict__ bc1,      // [H]
                                   const bf16* __restrict__ Wc2,       // [D][H]
                                   const float* __restrict__ bc2,      // [D]
                                   float* __restrict__ comp, int D, int H) {
  extern __shared__ __align__(32) bf16 smem[];   // 16 * D pred tile
  const int lane = threadIdx.x & 31;
  const int wave = threadIdx.x >> 5;
  const int half = lane >> 4;
  const int ln   = lane & 15;
  const int row0 = blockIdx.x * 16;

  __shared__ __align__(32) bf16 hbuf[16 * 256];
  __shared__ float rowacc[16];
  if (threadIdx.x < 16) rowacc[threadIdx.x] = 0.0f;

#if HAVE_TDM
  if (wave == 0) {
    tdm_load_2d(PredBf + (size_t)row0 * D, lds_offset(smem), 16, D, D);
    wait_tensor0();
  }
#else
  stage_tile(PredBf + (size_t)row0 * D, smem, 16, D, D, threadIdx.x, blockDim.x);
#endif
  __syncthreads();

  // phase 1: h = relu(pred @ Wc1^T + bc1) -> LDS (bf16)
  for (int nt = wave; nt < (H >> 4); nt += 8) {
    const int n0 = nt * 16;
    v8f acc = {};
    for (int k = 0; k < D; k += 32) {
      v16bf a = *(const v16bf*)(smem + ln * D + k + half * 16);    // ds_load
      v16bf b = *(const v16bf*)(Wc1 + (size_t)(n0 + ln) * D + k + half * 16);
      acc = wmma_bf16(a, b, acc);
    }
    const int n = n0 + ln;
    const float b1n = bc1[n];
#pragma unroll
    for (int r = 0; r < 8; ++r) {
      float v = acc[r] + b1n;
      v = v > 0.0f ? v : 0.0f;
      hbuf[(half * 8 + r) * 256 + n] = (bf16)v;
    }
  }
  __syncthreads();

  // phase 2: recon = h @ Wc2^T + bc2; accumulate (pred - recon)^2 per row
  float part[8];
#pragma unroll
  for (int r = 0; r < 8; ++r) part[r] = 0.0f;
  for (int nt = wave; nt < (D >> 4); nt += 8) {
    const int n0 = nt * 16;
    v8f acc = {};
    for (int k = 0; k < H; k += 32) {
      v16bf a = *(const v16bf*)(hbuf + ln * 256 + k + half * 16);  // ds_load
      v16bf b = *(const v16bf*)(Wc2 + (size_t)(n0 + ln) * H + k + half * 16);
      acc = wmma_bf16(a, b, acc);
    }
    const int n = n0 + ln;
    const float b2n = bc2[n];
#pragma unroll
    for (int r = 0; r < 8; ++r) {
      const int row = half * 8 + r;
      float recon = acc[r] + b2n;
      float diff = Pred[(size_t)(row0 + row) * D + n] - recon;
      part[r] += diff * diff;
    }
  }
#pragma unroll
  for (int r = 0; r < 8; ++r)
    atomicAdd(&rowacc[half * 8 + r], part[r]);
  __syncthreads();
  if (threadIdx.x < 16)
    comp[row0 + threadIdx.x] = rowacc[threadIdx.x] * (1.0f / 1024.0f);
}

// ---------------------------------------------------------------- finalize
__global__ void finalize_kernel(const float* __restrict__ diss,
                                const float* __restrict__ unc,
                                const float* __restrict__ ent,
                                const float* __restrict__ simmax,
                                const float* __restrict__ novn,
                                const float* __restrict__ comp,
                                float* __restrict__ out, int n) {
  int b = blockIdx.x * blockDim.x + threadIdx.x;
  if (b >= n) return;
  out[b * 4 + 0] = diss[b];
  out[b * 4 + 1] = unc[b] + 0.1f * ent[b];
  out[b * 4 + 2] = 0.7f * (1.0f - simmax[b]) + 0.3f * novn[b];
  out[b * 4 + 3] = comp[b];
}

// =========================================================================
extern "C" void kernel_launch(void* const* d_in, const int* in_sizes, int n_in,
                              void* d_out, int out_size, void* d_ws, size_t ws_size,
                              hipStream_t stream) {
  const int Bn = BROWS, Dn = DDIM;
  const float* pred   = (const float*)d_in[0];
  const float* act    = (const float*)d_in[1];
  const float* pmem   = (const float*)d_in[2];
  const float* W_d1   = (const float*)d_in[3];
  const float* b_d1   = (const float*)d_in[4];
  const float* W_d2   = (const float*)d_in[5];
  const float* b_d2   = (const float*)d_in[6];
  const float* W_u1   = (const float*)d_in[7];
  const float* b_u1   = (const float*)d_in[8];
  const float* W_u2   = (const float*)d_in[9];
  const float* b_u2   = (const float*)d_in[10];
  const float* W_n1   = (const float*)d_in[11];
  const float* b_n1   = (const float*)d_in[12];
  const float* W_n2   = (const float*)d_in[13];
  const float* b_n2   = (const float*)d_in[14];
  const float* W_c1   = (const float*)d_in[15];
  const float* b_c1   = (const float*)d_in[16];
  const float* W_c2   = (const float*)d_in[17];
  const float* b_c2   = (const float*)d_in[18];
  float* out = (float*)d_out;

  char* ws = (char*)d_ws;
  size_t off = 0;
  auto alloc = [&](size_t bytes) -> void* {
    void* p = ws + off;
    off = (off + bytes + 255) & ~(size_t)255;
    return p;
  };
  bf16* pred_bf = (bf16*)alloc((size_t)Bn * Dn * sizeof(bf16));
  bf16* act_bf  = (bf16*)alloc((size_t)Bn * Dn * sizeof(bf16));
  bf16* Wd1_bf  = (bf16*)alloc((size_t)1024 * 2048 * sizeof(bf16));
  bf16* Wu1_bf  = (bf16*)alloc((size_t)512 * 1024 * sizeof(bf16));
  bf16* Wn1_bf  = (bf16*)alloc((size_t)512 * 1024 * sizeof(bf16));
  bf16* Wc1_bf  = (bf16*)alloc((size_t)256 * 1024 * sizeof(bf16));
  bf16* Wc2_bf  = (bf16*)alloc((size_t)1024 * 256 * sizeof(bf16));
  bf16* mem_bf  = (bf16*)alloc((size_t)MPAD * Dn * sizeof(bf16));
  float* m_norm  = (float*)alloc(MPAD * sizeof(float));
  float* entropy = (float*)alloc((size_t)Bn * sizeof(float));
  float* a_norm  = (float*)alloc((size_t)Bn * sizeof(float));
  float* diss    = (float*)alloc((size_t)Bn * sizeof(float));
  float* unc     = (float*)alloc((size_t)Bn * sizeof(float));
  float* novn    = (float*)alloc((size_t)Bn * sizeof(float));
  float* simmax  = (float*)alloc((size_t)Bn * sizeof(float));
  float* comp    = (float*)alloc((size_t)Bn * sizeof(float));

  auto cvt = [&](const float* s, bf16* d, int n) {
    cvt_f32_to_bf16<<<(n + 255) / 256, 256, 0, stream>>>(s, d, n);
  };
  cvt(pred, pred_bf, Bn * Dn);
  cvt(act,  act_bf,  Bn * Dn);
  cvt(W_d1, Wd1_bf, 1024 * 2048);
  cvt(W_u1, Wu1_bf, 512 * 1024);
  cvt(W_n1, Wn1_bf, 512 * 1024);
  cvt(W_c1, Wc1_bf, 256 * 1024);
  cvt(W_c2, Wc2_bf, 1024 * 256);
  cvt_pad_mem<<<(MPAD * Dn + 255) / 256, 256, 0, stream>>>(pmem, mem_bf, MROWS, Dn, MPAD);

  mnorm_kernel<<<1, 128, 0, stream>>>(pmem, m_norm, MROWS, Dn, MPAD);
  row_stats_kernel<<<Bn, 256, 0, stream>>>(pred, act, entropy, a_norm, Dn);

  const int rowBlocks = Bn / 16;  // 1024
  // dissonance: X = [pred | actual], K = 2048, N = 1024 (64 KB LDS tile)
  fused_mlp1_kernel<<<rowBlocks, 256, 16 * 2048 * sizeof(bf16), stream>>>(
      pred_bf, act_bf, Wd1_bf, b_d1, W_d2, b_d2, diss, 1024, 1024, 1024);
  // uncertainty (neural part): pred, K = 1024, N = 512
  fused_mlp1_kernel<<<rowBlocks, 256, 16 * 1024 * sizeof(bf16), stream>>>(
      pred_bf, nullptr, Wu1_bf, b_u1, W_u2, b_u2, unc, 512, 1024, 0);
  // novelty (neural part): actual, K = 1024, N = 512
  fused_mlp1_kernel<<<rowBlocks, 256, 16 * 1024 * sizeof(bf16), stream>>>(
      act_bf, nullptr, Wn1_bf, b_n1, W_n2, b_n2, novn, 512, 1024, 0);

  novelty_sims_kernel<<<rowBlocks, 256, 16 * 1024 * sizeof(bf16), stream>>>(
      act_bf, mem_bf, a_norm, m_norm, simmax, Dn);
  compression_kernel<<<rowBlocks, 256, 16 * 1024 * sizeof(bf16), stream>>>(
      pred, pred_bf, Wc1_bf, b_c1, Wc2_bf, b_c2, comp, Dn, 256);

  finalize_kernel<<<(Bn + 255) / 256, 256, 0, stream>>>(diss, unc, entropy, simmax,
                                                        novn, comp, out, Bn);
}
